// Attention_Local_Predictive_58634893525781
// MI455X (gfx1250) — matrix operand order repaired and checked by reading, not compiled, not run on hardware
//
#include <hip/hip_runtime.h>
#include <hip/hip_bf16.h>

typedef __attribute__((ext_vector_type(16))) __bf16 v16bf;
typedef __attribute__((ext_vector_type(8)))  __bf16 v8bf;
typedef __attribute__((ext_vector_type(4)))  __bf16 v4bf;
typedef __attribute__((ext_vector_type(8)))  float  v8f;
typedef __attribute__((ext_vector_type(4)))  unsigned int v4u;
typedef __attribute__((ext_vector_type(8)))  int v8i;
typedef __attribute__((ext_vector_type(4)))  int v4i;

constexpr int kB = 8;
constexpr int kE = 4096;
constexpr int kD = 1024;
constexpr int kH = 512;
constexpr float kSigma = 5.0f;

#if __has_builtin(__builtin_amdgcn_tensor_load_to_lds) && \
    __has_builtin(__builtin_amdgcn_s_wait_tensorcnt)
#define USE_TDM 1
#else
#define USE_TDM 0
#endif

// ---------------------------------------------------------------------------
// A-fragment (16x32 MxK bf16) built from an f32 row in global memory.
// Lane holds row m; elements 0..15 -> K=kb+0..15 (contiguous),
// caller passes kb including (lane>=16) ? 16 : 0.
__device__ __forceinline__ v16bf ldA_f32row(const float* rowp, int kb) {
  const float4* p = (const float4*)(rowp + kb);
  const float4* q = (const float4*)(rowp + kb + 8);
  float4 a = p[0], b = p[1], c = q[0], d = q[1];
  v16bf f;
  f[0] = (__bf16)a.x; f[1] = (__bf16)a.y; f[2]  = (__bf16)a.z; f[3]  = (__bf16)a.w;
  f[4] = (__bf16)b.x; f[5] = (__bf16)b.y; f[6]  = (__bf16)b.z; f[7]  = (__bf16)b.w;
  f[8] = (__bf16)c.x; f[9] = (__bf16)c.y; f[10] = (__bf16)c.z; f[11] = (__bf16)c.w;
  f[12] = (__bf16)d.x; f[13] = (__bf16)d.y; f[14] = (__bf16)d.z; f[15] = (__bf16)d.w;
  return f;
}

#if USE_TDM
// One 2D TDM tile load: 32 rows x 512 bf16, row stride 512, into LDS.
__device__ __forceinline__ void tdm_load_panel(unsigned ldsOff,
                                               unsigned long long ga) {
  v4u g0;
  g0[0] = 1u;                                   // count=1, user mode
  g0[1] = ldsOff;                               // D#.lds_addr (bytes)
  g0[2] = (unsigned)(ga & 0xffffffffu);         // global_addr[31:0]
  g0[3] = (unsigned)((ga >> 32) & 0x01ffffffu)  // global_addr[56:32]
          | (2u << 30);                         // type = 2 ("image")
  v8i g1;
  g1[0] = 0x00010000;          // data_size=1 (2B), no mcast/pad/iterate
  g1[1] = (int)(512u << 16);   // tensor_dim0 = 512
  g1[2] = 0x10000000;          // tensor_dim1.lo = 4096
  g1[3] = 0x02000000;          // tile_dim0 = 512
  g1[4] = 32;                  // tile_dim1 = 32, tile_dim2 = 0
  g1[5] = 512;                 // tensor_dim0_stride = 512
  g1[6] = 0;
  g1[7] = 0;
  v4i z4 = {0, 0, 0, 0};
  v8i z8 = {0, 0, 0, 0, 0, 0, 0, 0};
  __builtin_amdgcn_tensor_load_to_lds(g0, g1, z4, z4, z8, 0);
}
#endif

// ---------------------------------------------------------------------------
// Kernel 0a: enc (f32) -> encB (bf16), straight copy (flip handled by tile
// addressing in the main kernel).
__global__ __launch_bounds__(256) void conv_enc(const float* __restrict__ enc,
                                                __bf16* __restrict__ encB) {
  const size_t i = (size_t)blockIdx.x * 256 + threadIdx.x;   // float4 index
  float4 f = ((const float4*)enc)[i];
  v4bf o;
  o[0] = (__bf16)f.x; o[1] = (__bf16)f.y; o[2] = (__bf16)f.z; o[3] = (__bf16)f.w;
  *(v4bf*)(encB + i * 4) = o;
}

// Kernel 0b: W (f32, [k][n]) -> WBt (bf16, [n][k]) transposed.
__global__ __launch_bounds__(128) void conv_w(const float* __restrict__ W,
                                              __bf16* __restrict__ WBt) {
  const int k = blockIdx.x;
  for (int n = threadIdx.x; n < kH; n += 128)
    WBt[(size_t)n * kH + k] = (__bf16)W[(size_t)k * kH + n];
}

// ---------------------------------------------------------------------------
// Kernel 1: pos[b,d] = sigmoid( tanh(dec·W) · v ) * en_len[b]
__global__ __launch_bounds__(32) void pos_kernel(
    const float* __restrict__ dec, const __bf16* __restrict__ WBt,
    const float* __restrict__ v, const float* __restrict__ en_len,
    float* __restrict__ pos) {
  const int lane = threadIdx.x;
  const int b  = blockIdx.x >> 6;
  const int d0 = (blockIdx.x & 63) << 4;
  const int n15 = lane & 15;
  const int klo = (lane & 16) ? 8 : 0;
  const int kB16 = (lane & 16) ? 16 : 0;

  const float* decRow = dec + ((size_t)b * kD + d0 + n15) * kH;
  v16bf afr[16];
#pragma unroll
  for (int kc = 0; kc < 16; ++kc) afr[kc] = ldA_f32row(decRow, kc * 32 + kB16);

  float rs[8] = {0.f, 0.f, 0.f, 0.f, 0.f, 0.f, 0.f, 0.f};
  for (int nt = 0; nt < 32; ++nt) {
    const int n = nt * 16 + n15;
    const __bf16* wrow = WBt + (size_t)n * kH;
    v8f acc = {};
#pragma unroll
    for (int kc = 0; kc < 16; ++kc) {
      v16bf bfr = *(const v16bf*)(wrow + kc * 32 + kB16);
      acc = __builtin_amdgcn_wmma_f32_16x16x32_bf16(
          false, afr[kc], false, bfr, (short)0, acc, false, false);
    }
    const float vn = v[n];
#pragma unroll
    for (int r = 0; r < 8; ++r) rs[r] += tanhf(acc[r]) * vn;
  }

#pragma unroll
  for (int r = 0; r < 8; ++r) {
#pragma unroll
    for (int off = 1; off < 16; off <<= 1) rs[r] += __shfl_xor(rs[r], off, 32);
  }
  if (n15 == 0) {
    const float el = en_len[b];
    const int mhi = (lane >> 4) * 8;
#pragma unroll
    for (int r = 0; r < 8; ++r) {
      const float s = 1.0f / (1.0f + __expf(-rs[r]));
      pos[(size_t)b * kD + d0 + mhi + r] = s * el;
    }
  }
}

// ---------------------------------------------------------------------------
// Kernel 2: fused local-predictive attention.
// Workgroup = 8 waves = one (b, 32 d-rows) tile: wave = (dsub, q) with
// dsub = d-subtile of 16 rows, q = h-quarter. Double-buffered TDM panels
// (32 flipped-order rows each); flipped index e' <-> panel row (31 - e').
__global__ __launch_bounds__(256) void attn_main(
    const __bf16* __restrict__ encB, const float* __restrict__ dec,
    const float* __restrict__ pos, float* __restrict__ out) {
  __shared__ __align__(64) __bf16 encS[2][32][kH];       // 64 KB, 2 panels
  __shared__ __align__(64) float  scoreS[2][4][16][32];  // 16 KB partials
  __shared__ __align__(64) __bf16 sgS[2][16][32];        //  2 KB score*gauss

  const int t    = threadIdx.x;
  const int wave = t >> 5;
  const int dsub = wave >> 2;               // 0..1: d-subtile
  const int q    = wave & 3;                // 0..3: h-quarter
  const int lane = t & 31;
  const int n15  = lane & 15;
  const int mhi  = (lane >> 4) * 8;
  const int klo  = (lane & 16) ? 8 : 0;     // unused; A also uses kB16
  const int kB16 = (lane & 16) ? 16 : 0;    // B pattern
  const int b  = blockIdx.x >> 5;           // kD/32 = 32 tiles per batch
  const int d0 = (blockIdx.x & 31) << 5;

  // dec A-fragments for this wave's (d-subtile, K-quarter)
  const float* decRow = dec + ((size_t)b * kD + d0 + dsub * 16 + n15) * kH;
  v16bf adec[4];
#pragma unroll
  for (int kc = 0; kc < 4; ++kc)
    adec[kc] = ldA_f32row(decRow, q * 128 + kc * 32 + kB16);

  // reduction-phase mapping: thread t -> (dsub2, row dr, 4 cols from c0)
  const int dsub2 = t >> 7;
  const int dr = (t >> 3) & 15;
  const int c0 = (t & 7) * 4;
  const float pcenter = pos[(size_t)b * kD + d0 + dsub2 * 16 + dr];

  v8f accO[8];
#pragma unroll
  for (int i = 0; i < 8; ++i) accO[i] = (v8f){};

  const unsigned ldsOff0 = (unsigned)(size_t)&encS[0][0][0];
  const unsigned long long encAddr = (unsigned long long)(size_t)encB;
  const size_t batchRow = (size_t)b * kE;
  // panel p (p = e0/32) covers original rows [kE-32-e0, kE-e0)
  auto panel_ga = [&](int e0) -> unsigned long long {
    return encAddr + (batchRow + (size_t)(kE - 32 - e0)) * kH * 2ull;
  };

#if USE_TDM
  if (t < 32) tdm_load_panel(ldsOff0, panel_ga(0));   // prologue: panel 0
#endif

  for (int e0 = 0; e0 < kE; e0 += 32) {
    const int cur = (e0 >> 5) & 1;
#if USE_TDM
    if (t < 32) {     // prefetch next panel, then wait for current (in-order)
      if (e0 + 32 < kE) {
        tdm_load_panel(ldsOff0 + (unsigned)((cur ^ 1) * 32 * kH * 2),
                       panel_ga(e0 + 32));
        __builtin_amdgcn_s_wait_tensorcnt(1);
      } else {
        __builtin_amdgcn_s_wait_tensorcnt(0);
      }
    }
#else
    for (int i = t; i < 32 * kH / 8; i += 256) {
      const int row = i >> 6, c8 = (i & 63) * 8;
      *(v8bf*)(&encS[cur][row][c8]) =
          *(const v8bf*)(encB + (batchRow + (kE - 32 - e0) + row) * kH + c8);
    }
#endif
    __syncthreads();
    const __bf16* pan = &encS[cur][0][0];

    // ---- score partial over this wave's K-quarter: S[dsub][16,32] --------
    // B[k=h][n=e'] = encF[e'][h] = pan[31-e'][h]  (contiguous-K row reads)
    v8f s0 = {}, s1 = {};
#pragma unroll
    for (int kc = 0; kc < 4; ++kc) {
      const int kb = q * 128 + kc * 32 + kB16;
      v16bf b0 = *(const v16bf*)(pan + (size_t)(31 - n15) * kH + kb);
      v16bf b1 = *(const v16bf*)(pan + (size_t)(15 - n15) * kH + kb);
      s0 = __builtin_amdgcn_wmma_f32_16x16x32_bf16(false, adec[kc], false, b0,
                                                   (short)0, s0, false, false);
      s1 = __builtin_amdgcn_wmma_f32_16x16x32_bf16(false, adec[kc], false, b1,
                                                   (short)0, s1, false, false);
    }
#pragma unroll
    for (int r = 0; r < 8; ++r) {
      scoreS[dsub][q][mhi + r][n15]      = s0[r];
      scoreS[dsub][q][mhi + r][16 + n15] = s1[r];
    }
    __syncthreads();

    // ---- cross-wave reduce + gaussian window -> sgS (bf16) ---------------
#pragma unroll
    for (int cc = 0; cc < 4; ++cc) {
      const int c = c0 + cc;
      float s = scoreS[dsub2][0][dr][c] + scoreS[dsub2][1][dr][c] +
                scoreS[dsub2][2][dr][c] + scoreS[dsub2][3][dr][c];
      const float z = ((float)(e0 + c) - pcenter) * (1.0f / kSigma);
      s *= __expf(-0.5f * z * z);
      sgS[dsub2][dr][c] = (__bf16)s;
    }
    __syncthreads();

    // ---- out[16, h-quarter] += sg[16,32] x encF[32, h-quarter] -----------
    v16bf asg;
    {
      v8bf lo = *(const v8bf*)(&sgS[dsub][n15][kB16]);
      v8bf hi = *(const v8bf*)(&sgS[dsub][n15][kB16 + 8]);
#pragma unroll
      for (int i = 0; i < 8; ++i) { asg[i] = lo[i]; asg[8 + i] = hi[i]; }
    }
#pragma unroll
    for (int nt = 0; nt < 8; ++nt) {
      const int h = q * 128 + nt * 16 + n15;
      v16bf bfr;                     // B[k=e'][n=h] = pan[31-e'][h]
#pragma unroll
      for (int i = 0; i < 16; ++i) bfr[i] = pan[(size_t)(31 - (kB16 + i)) * kH + h];
      accO[nt] = __builtin_amdgcn_wmma_f32_16x16x32_bf16(
          false, asg, false, bfr, (short)0, accO[nt], false, false);
    }
    __syncthreads();
  }

  // ---- write out tile (f32) ----------------------------------------------
  float* outB = out + ((size_t)b * kD + d0 + dsub * 16) * kH;
#pragma unroll
  for (int nt = 0; nt < 8; ++nt) {
    const int h = q * 128 + nt * 16 + n15;
#pragma unroll
    for (int r = 0; r < 8; ++r)
      outB[(size_t)(mhi + r) * kH + h] = accO[nt][r];
  }
}

// ---------------------------------------------------------------------------
extern "C" void kernel_launch(void* const* d_in, const int* in_sizes, int n_in,
                              void* d_out, int out_size, void* d_ws, size_t ws_size,
                              hipStream_t stream) {
  const float* enc    = (const float*)d_in[0];
  const float* dec    = (const float*)d_in[1];
  const float* en_len = (const float*)d_in[2];
  const float* W      = (const float*)d_in[3];
  const float* v      = (const float*)d_in[4];
  float* out = (float*)d_out;

  char* ws = (char*)d_ws;
  float*  pos  = (float*)ws;                                   // 32 KB
  __bf16* encB = (__bf16*)(ws + (32 * 1024));                  // 32 MB
  __bf16* WBt  = (__bf16*)(ws + (32 * 1024) + (size_t)kB * kE * kH * 2);

  conv_enc<<<dim3(kB * kE * kH / 4 / 256), dim3(256), 0, stream>>>(enc, encB);
  conv_w  <<<dim3(kH), dim3(128), 0, stream>>>(W, WBt);

  pos_kernel<<<dim3(kB * kD / 16), dim3(32),  0, stream>>>(dec, WBt, v, en_len, pos);
  attn_main <<<dim3(kB * kD / 32), dim3(256), 0, stream>>>(encB, dec, pos, out);
}
